// maxpooler_ring_79585743994944
// MI455X (gfx1250) — matrix operand order repaired and checked by reading, not compiled, not run on hardware
//
#include <hip/hip_runtime.h>
#include <math.h>

typedef __attribute__((ext_vector_type(16))) _Float16 v16h;
typedef __attribute__((ext_vector_type(8)))  float    v8f;

#define BB    32
#define NN    8192
#define DIN   64
#define DOUT  128
#define NRING 4
#define EPS   1e-5f
#define CHUNK 512                  // points per workgroup
#define NEG_INF -3.402823466e38f
#define MASK_NEG -1.0e30f          // C-operand penalty for wrong-ring columns

// workspace layout (bytes)
#define WS_A_OFF    0        // 64 (tile,khalf) blocks * 1024B  = 65536
#define WS_BIAS_OFF 65536    // 512 floats                      = 2048
#define WS_M_OFF    67584    // 32*512 unsigned max-keys        = 65536

// LDS layout (bytes) for K1
#define L_B    0         // 65536 : packed f16 B blocks (ring-sorted columns)
#define L_RS   65536     // 2048  : sorted ring labels
#define L_POS  67584     // 2048  : original point -> sorted column
#define L_M    69632     // 2048  : 512 max keys
#define L_CNT  71680     // 64    : [0..3]=counts, [4..7]=running offsets/ends
#define L_SIZE 71744

// order-preserving float <-> uint key (negatives work with integer atomicMax)
__device__ __forceinline__ unsigned fkey(float f) {
    unsigned b = __float_as_uint(f);
    return (b & 0x80000000u) ? ~b : (b | 0x80000000u);
}
__device__ __forceinline__ float fdec(unsigned u) {
    unsigned b = (u & 0x80000000u) ? (u & 0x7FFFFFFFu) : ~u;
    return __uint_as_float(b);
}

// ---------------------------------------------------------------------------
// K0: fold BN into W -> f16 A-matrix pack; folded bias; init global max keys.
// A layout (16x32 MxK): g=lane/16, M=lane%16; element e: v=e/2,h=e%2,
//   K = (v>=4 ? 16 : 0) + 8*g + 2*(v%4) + h
// ---------------------------------------------------------------------------
__global__ __launch_bounds__(256) void prep_kernel(
    const float* __restrict__ W, const float* __restrict__ bvec,
    const float* __restrict__ gamma, const float* __restrict__ beta,
    const float* __restrict__ mean, const float* __restrict__ var,
    _Float16* __restrict__ wsA, float* __restrict__ bias2,
    unsigned* __restrict__ Mg) {
    const int gtid = blockIdx.x * 256 + threadIdx.x;   // grid 64 -> 16384

    if (gtid < BB * NRING * DOUT) Mg[gtid] = 0u;       // below every valid key

    if (gtid < NRING * DOUT) {
        float sc = gamma[gtid] * rsqrtf(var[gtid] + EPS);
        bias2[gtid] = (bvec[gtid] - mean[gtid]) * sc + beta[gtid];
    }

    if (gtid < 64 * 32) {
        const int pair = gtid >> 5;                    // ct*2 + s
        const int lane = gtid & 31;
        const int t = pair >> 1, s = pair & 1;
        const int g = lane >> 4, M = lane & 15;
        const int r = t >> 3;
        const int o = (t & 7) * 16 + M;
        const float sc = gamma[r * DOUT + o] * rsqrtf(var[r * DOUT + o] + EPS);
        _Float16* dst = wsA + (size_t)pair * 512 + lane * 16;
        #pragma unroll
        for (int e = 0; e < 16; ++e) {
            const int v = e >> 1, h = e & 1;
            const int K = ((v >= 4) ? 16 : 0) + 8 * g + 2 * (v & 3) + h;
            dst[e] = (_Float16)(W[((size_t)r * DOUT + o) * DIN + s * 32 + K] * sc);
        }
    }
}

// ---------------------------------------------------------------------------
// K1: per (batch, 512-point chunk).
//   1) ring-histogram + prefix + rank: sorted column for every point
//   2) stage x chunk -> packed f16 B layout at ring-sorted columns
//   3) wave w serves ring r=w>>1, channel tiles r*8+(w&1)*4 .. +3 (A in regs),
//      sweeps only its ring's contiguous tile range; wrong-ring columns are
//      masked through the WMMA C operand (-1e30); raw v_max_num accumulate
//   4) one shuffle-reduce + LDS/global atomic-max epilogue
// ---------------------------------------------------------------------------
__global__ __launch_bounds__(256) void ringmax_wmma_kernel(
    const float* __restrict__ x, const int* __restrict__ ring,
    const _Float16* __restrict__ wsA, unsigned* __restrict__ Mg) {
    extern __shared__ char smem[];
    _Float16* sB   = (_Float16*)(smem + L_B);
    int*      sRs  = (int*)(smem + L_RS);
    int*      sPos = (int*)(smem + L_POS);
    unsigned* sM   = (unsigned*)(smem + L_M);
    int*      sCnt = (int*)(smem + L_CNT);
    const int tid  = threadIdx.x;
    const int wave = tid >> 5, lane = tid & 31;
    const int b     = blockIdx.y;
    const int chunk = blockIdx.x * CHUNK;

    if (tid < 8) sCnt[tid] = 0;
    for (int i = tid; i < 512; i += 256) sM[i] = 0u;
    __syncthreads();

    // phase 1: load ring labels, histogram (sPos used as temp label store)
    for (int i = tid; i < 512; i += 256) {
        const int rn = ring[(size_t)b * NN + chunk + i];
        sPos[i] = rn;
        atomicAdd(&sCnt[rn], 1);
    }
    __syncthreads();

    // phase 2: exclusive prefix -> running offsets in sCnt[4..7]
    if (tid == 0) {
        int run = 0;
        #pragma unroll
        for (int r = 0; r < NRING; ++r) { sCnt[4 + r] = run; run += sCnt[r]; }
    }
    __syncthreads();

    // phase 3: rank assignment; sPos = sorted column, sRs = sorted labels
    for (int i = tid; i < 512; i += 256) {
        const int rn  = sPos[i];
        const int pos = atomicAdd(&sCnt[4 + rn], 1);
        sPos[i] = pos;
        sRs[pos] = rn;
    }
    __syncthreads();   // sCnt[4+r] now holds END of ring r's sorted range

    // stage x chunk -> packed B layout at sorted columns (b32 packed stores)
    {
        const float* xb = x + (size_t)b * DIN * NN + chunk;
        #pragma unroll 4
        for (int it = 0; it < 16; ++it) {
            const int idx = tid + it * 256;          // 0..4095
            const int kp  = idx >> 7;                // 0..31 (k-pair)
            const int n4  = idx & 127;               // float4 column group
            const int k0  = kp * 2;
            const float4 r0 = *(const float4*)(xb + (size_t)k0 * NN + n4 * 4);
            const float4 r1 = *(const float4*)(xb + (size_t)(k0 + 1) * NN + n4 * 4);
            const int4  p4 = *(const int4*)(sPos + n4 * 4);
            const int s   = k0 >> 5;
            const int lhi = (k0 >> 3) & 1;
            const int e0  = ((k0 >> 4) & 1) * 8 + (k0 & 7);
            const float a0[4] = { r0.x, r0.y, r0.z, r0.w };
            const float a1[4] = { r1.x, r1.y, r1.z, r1.w };
            const int   pp[4] = { p4.x, p4.y, p4.z, p4.w };
            #pragma unroll
            for (int q = 0; q < 4; ++q) {
                const int pos = pp[q];
                _Float16* dst = sB + (size_t)((pos >> 4) * 2 + s) * 512
                                   + (lhi * 16 + (pos & 15)) * 16 + e0;
                unsigned short u0 = __builtin_bit_cast(unsigned short, (_Float16)a0[q]);
                unsigned short u1 = __builtin_bit_cast(unsigned short, (_Float16)a1[q]);
                *(unsigned*)dst = (unsigned)u0 | ((unsigned)u1 << 16);
            }
        }
    }
    __syncthreads();

    // this wave's ring and 4 channel tiles; A matrices live in registers
    const int r   = wave >> 1;
    const int ct0 = r * 8 + (wave & 1) * 4;
    v16h A[4][2];
    #pragma unroll
    for (int i = 0; i < 4; ++i)
        #pragma unroll
        for (int s = 0; s < 2; ++s)
            A[i][s] = *(const v16h*)(wsA + (size_t)((ct0 + i) * 2 + s) * 512
                                         + lane * 16);

    // wave-uniform sorted tile range for this ring
    int pEnd   = sCnt[4 + r];
    int pStart = (r == 0) ? 0 : sCnt[4 + r - 1];
    pStart = __builtin_amdgcn_readfirstlane(pStart);
    pEnd   = __builtin_amdgcn_readfirstlane(pEnd);
    const int t0 = pStart >> 4;
    const int t1 = (pEnd + 15) >> 4;

    const int col = lane & 15;
    float acc[4][8];
    #pragma unroll
    for (int i = 0; i < 4; ++i)
        #pragma unroll
        for (int j = 0; j < 8; ++j) acc[i][j] = NEG_INF;

    for (int pt = t0; pt < t1; ++pt) {
        const int rp = sRs[pt * 16 + col];
        const float m = (rp == r) ? 0.0f : MASK_NEG;   // per-column C seed
        v8f cinit;
        #pragma unroll
        for (int j = 0; j < 8; ++j) cinit[j] = m;
        v16h b0 = *(const v16h*)(sB + (size_t)(pt * 2 + 0) * 512 + lane * 16);
        v16h b1 = *(const v16h*)(sB + (size_t)(pt * 2 + 1) * 512 + lane * 16);
        #pragma unroll
        for (int i = 0; i < 4; ++i) {
            v8f c = __builtin_amdgcn_wmma_f32_16x16x32_f16(
                false, A[i][0], false, b0, (short)0, cinit, false, false);
            c = __builtin_amdgcn_wmma_f32_16x16x32_f16(
                false, A[i][1], false, b1, (short)0, c, false, false);
            #pragma unroll
            for (int j = 0; j < 8; ++j)
                asm("v_max_num_f32 %0, %0, %1"
                    : "+v"(acc[i][j]) : "v"(c[j]));
        }
    }

    // epilogue: reduce across the 16 point columns once, merge into sM
    #pragma unroll
    for (int i = 0; i < 4; ++i) {
        const int ct = ct0 + i;
        #pragma unroll
        for (int j = 0; j < 8; ++j) {
            float v = acc[i][j];
            v = fmaxf(v, __shfl_xor(v, 1, 16));
            v = fmaxf(v, __shfl_xor(v, 2, 16));
            v = fmaxf(v, __shfl_xor(v, 4, 16));
            v = fmaxf(v, __shfl_xor(v, 8, 16));
            if (col == 0) {   // lanes 0 and 16 commit rows M=j and M=j+8
                const int idx = r * DOUT + (ct & 7) * 16 + j + 8 * (lane >> 4);
                atomicMax(&sM[idx], fkey(v));
            }
        }
    }
    __syncthreads();
    for (int i = tid; i < 512; i += 256)
        atomicMax(&Mg[(size_t)b * 512 + i], sM[i]);
}

// ---------------------------------------------------------------------------
// K2: scatter. out[b][o][n] = decode(M[b][ring[b,n]][o]) + bias2. float4
// stores, coalesced across threads per channel row.
// ---------------------------------------------------------------------------
__global__ __launch_bounds__(256) void scatter_kernel(
    const int* __restrict__ ring, const unsigned* __restrict__ Mg,
    const float* __restrict__ bias2, float* __restrict__ out) {
    __shared__ float sV[NRING * DOUT];
    const int tid = threadIdx.x;
    const int b   = blockIdx.y;
    for (int i = tid; i < NRING * DOUT; i += 256)
        sV[i] = fdec(Mg[(size_t)b * 512 + i]) + bias2[i];
    __syncthreads();

    const int n  = blockIdx.x * 1024 + tid * 4;
    const int r0 = ring[(size_t)b * NN + n + 0] * DOUT;
    const int r1 = ring[(size_t)b * NN + n + 1] * DOUT;
    const int r2 = ring[(size_t)b * NN + n + 2] * DOUT;
    const int r3 = ring[(size_t)b * NN + n + 3] * DOUT;
    float* ob = out + (size_t)b * DOUT * NN + n;
    #pragma unroll 4
    for (int o = 0; o < DOUT; ++o) {
        float4 vv = { sV[r0 + o], sV[r1 + o], sV[r2 + o], sV[r3 + o] };
        *(float4*)(ob + (size_t)o * NN) = vv;
    }
}

extern "C" void kernel_launch(void* const* d_in, const int* in_sizes, int n_in,
                              void* d_out, int out_size, void* d_ws, size_t ws_size,
                              hipStream_t stream) {
    const float* x     = (const float*)d_in[0];
    const int*   ring  = (const int*)d_in[1];
    const float* W     = (const float*)d_in[2];
    const float* bvec  = (const float*)d_in[3];
    const float* gamma = (const float*)d_in[4];
    const float* beta  = (const float*)d_in[5];
    const float* mean  = (const float*)d_in[6];
    const float* var   = (const float*)d_in[7];
    (void)in_sizes; (void)n_in; (void)out_size; (void)ws_size;

    char* ws = (char*)d_ws;
    _Float16* wsA   = (_Float16*)(ws + WS_A_OFF);
    float*    bias2 = (float*)(ws + WS_BIAS_OFF);
    unsigned* Mg    = (unsigned*)(ws + WS_M_OFF);
    float*    out   = (float*)d_out;

    prep_kernel<<<dim3(64), dim3(256), 0, stream>>>(W, bvec, gamma, beta, mean,
                                                    var, wsA, bias2, Mg);
    ringmax_wmma_kernel<<<dim3(16, 32), dim3(256), L_SIZE, stream>>>(
        x, ring, wsA, Mg);
    scatter_kernel<<<dim3(8, 32), dim3(256), 0, stream>>>(ring, Mg, bias2, out);
}
